// InterpAttentionKHeadsNet_2499670966531
// MI455X (gfx1250) — matrix operand order repaired and checked by reading, not compiled, not run on hardware
//
#include <hip/hip_runtime.h>

typedef __attribute__((ext_vector_type(16))) __bf16          v16bf;
typedef __attribute__((ext_vector_type(8)))  float           v8f;
typedef __attribute__((ext_vector_type(4)))  float           v4f;
typedef __attribute__((ext_vector_type(8)))  unsigned short  v8u16;

union BF16x16 { v16bf bv; v8u16 h[2]; };

// ---------- bf16 <-> f32 (round-to-nearest-even) ----------
__device__ __forceinline__ unsigned short f2bf(float f) {
    union { float f; unsigned u; } x; x.f = f;
    unsigned r = x.u + 0x7FFFu + ((x.u >> 16) & 1u);
    return (unsigned short)(r >> 16);
}
__device__ __forceinline__ float bf2f(unsigned short h) {
    union { unsigned u; float f; } x; x.u = ((unsigned)h) << 16;
    return x.f;
}

// ---------- problem constants ----------
#define LATENT 256
#define KNN    16
#define HEADS  64
#define TQ     8            // queries per workgroup
#define NCOL   128          // TQ * KNN columns per workgroup
#define K1PAD  288          // 259 padded up to multiple of 32

// ---------- LDS layout (bytes) ----------
#define S_ACTA 0                        // ushort [NCOL][288] col-major  -> 73728 B
#define S_ACTB 73728                    // ushort [NCOL][256] col-major  -> 65536 B
#define S_M    139264                   // float  [64][8]                ->  2048 B
#define S_Z    141312                   // float  [64][8]                ->  2048 B
#define S_ATT  143360                   // float  [NCOL]                 ->   512 B
#define S_S    143872                   // float  [NCOL]                 ->   512 B
#define SMEM_BYTES 144384
// QBUF (float [64][NCOL] = 32768 B) overlays S_ACTA after layer 3.

// =====================================================================
// Pre-pass 1: latents [B,256,N] f32 -> lat_bf [B][N][256] bf16 (contiguous per point)
// =====================================================================
__global__ void prep_latents(const float* __restrict__ latents,
                             unsigned short* __restrict__ lat_bf, int N) {
    int n = blockIdx.x * 64 + threadIdx.x;
    int b = blockIdx.y;
    if (n >= N) return;
    const float* src = latents + (size_t)b * LATENT * N + n;
    unsigned short* dst = lat_bf + ((size_t)b * N + n) * LATENT;
    for (int c0 = 0; c0 < LATENT; c0 += 8) {
        v8u16 o;
#pragma unroll
        for (int j = 0; j < 8; j++) o[j] = f2bf(src[(size_t)(c0 + j) * N]);
        *(v8u16*)(dst + c0) = o;
    }
}

// =====================================================================
// Pre-pass 2: weights -> bf16 (W1 padded 259->288), wfold = W8 @ Wv, cfold = W8 . bv
// =====================================================================
__global__ void prep_weights(const float* __restrict__ W1, const float* __restrict__ W2,
                             const float* __restrict__ W3, const float* __restrict__ Wq,
                             const float* __restrict__ Wv, const float* __restrict__ W8,
                             const float* __restrict__ bv,
                             unsigned short* __restrict__ w1p, unsigned short* __restrict__ w2b,
                             unsigned short* __restrict__ w3b, unsigned short* __restrict__ wqb,
                             float* __restrict__ wfold) {
    const int T1 = 256 * K1PAD;            // 73728
    const int T2 = T1 + 256 * 256;         // 139264
    const int T3 = T2 + 256 * 256;         // 204800
    const int T4 = T3 + 64 * 256;          // 221184
    const int T5 = T4 + 257;               // 221441
    int idx = blockIdx.x * blockDim.x + threadIdx.x;
    if (idx < T1) {
        int o = idx / K1PAD, k = idx % K1PAD;
        float v = (k < 259) ? W1[o * 259 + k] : 0.0f;
        w1p[idx] = f2bf(v);
    } else if (idx < T2) {
        int i = idx - T1; w2b[i] = f2bf(W2[i]);
    } else if (idx < T3) {
        int i = idx - T2; w3b[i] = f2bf(W3[i]);
    } else if (idx < T4) {
        int i = idx - T3; wqb[i] = f2bf(Wq[i]);
    } else if (idx < T5) {
        int i = idx - T4;
        if (i < 256) {
            float a = 0.0f;
            for (int o = 0; o < 256; o++) a += W8[o] * Wv[o * 256 + i];
            wfold[i] = a;
        } else {
            float a = 0.0f;
            for (int o = 0; o < 256; o++) a += W8[o] * bv[o];
            wfold[256] = a;
        }
    }
}

// =====================================================================
// WMMA GEMM over one LDS activation panel.
//   A (weights, global bf16, row-major [Mrows x KDIM])
//   B (activations, LDS bf16, column-major [NCOL][INSTRIDE])
//   out: bf16 LDS col-major (bias+ReLU) or f32 row-major qbuf (bias only)
// =====================================================================
template <int KDIM, int INSTRIDE, int OUTSTRIDE, int MT, bool RELU, bool F32OUT>
__device__ __forceinline__ void gemm_panel(const unsigned short* __restrict__ Wb,
                                           const float* __restrict__ bias,
                                           const unsigned short* actin,
                                           unsigned short* actout,
                                           float* qout,
                                           int wave, int lane) {
    const int lhalf = lane & 15;
    const int hi    = lane >> 4;           // 0 or 1
    for (int t = wave; t < MT * 8; t += 8) {
        const int mt = t >> 3;
        const int nt = t & 7;
        const int row = mt * 16 + lhalf;   // A row this lane loads
        const int col = nt * 16 + lhalf;   // B column this lane loads
        v8f acc = {};
        const unsigned short* wrow = Wb + (size_t)row * KDIM + hi * 8;
        const unsigned short* cin  = actin + (size_t)col * INSTRIDE + hi * 8;
#pragma unroll
        for (int kk = 0; kk < KDIM; kk += 32) {
            BF16x16 a, bfr;
            // A 16x32 bf16: lanes 0-15 hold K {kk..kk+7, kk+16..kk+23}, lanes 16-31 the +8 halves
            a.h[0] = *(const v8u16*)(wrow + kk);
            a.h[1] = *(const v8u16*)(wrow + kk + 16);
            // B 32x16 bf16: lanes 0-15 hold K {kk..kk+7, kk+16..kk+23} of col, lanes 16-31 the +8 halves
            bfr.h[0] = *(const v8u16*)(cin + kk);
            bfr.h[1] = *(const v8u16*)(cin + kk + 16);
            acc = __builtin_amdgcn_wmma_f32_16x16x32_bf16(
                false, a.bv, false, bfr.bv, (short)0, acc, false, false);
        }
        // C/D layout: VGPR r, lanes 0-15 -> M = mt*16 + r, lanes 16-31 -> M = mt*16 + 8 + r
        const int orow = mt * 16 + hi * 8;
        v4f b0 = *(const v4f*)(bias + orow);
        v4f b1 = *(const v4f*)(bias + orow + 4);
        float bb[8] = {b0[0], b0[1], b0[2], b0[3], b1[0], b1[1], b1[2], b1[3]};
        if constexpr (F32OUT) {
#pragma unroll
            for (int r = 0; r < 8; r++)
                qout[(orow + r) * NCOL + col] = acc[r] + bb[r];
        } else {
            v8u16 o;
#pragma unroll
            for (int r = 0; r < 8; r++) {
                float v = acc[r] + bb[r];
                if (RELU) v = fmaxf(v, 0.0f);
                o[r] = f2bf(v);
            }
            *(v8u16*)(actout + (size_t)col * OUTSTRIDE + orow) = o;
        }
    }
}

// =====================================================================
// Main fused kernel: one workgroup = 8 queries (128 columns)
// =====================================================================
__global__ void __launch_bounds__(256)
interp_attention_main(const unsigned short* __restrict__ lat_bf,
                      const float* __restrict__ pts,
                      const float* __restrict__ ptsq,
                      const int* __restrict__ proj,
                      const unsigned short* __restrict__ w1p, const float* __restrict__ b1,
                      const unsigned short* __restrict__ w2b, const float* __restrict__ b2,
                      const unsigned short* __restrict__ w3b, const float* __restrict__ b3,
                      const unsigned short* __restrict__ wqb, const float* __restrict__ bq,
                      const float* __restrict__ wfold, const float* __restrict__ b8,
                      float* __restrict__ out, int N, int NQ) {
    __shared__ __align__(16) unsigned char smem[SMEM_BYTES];
    unsigned short* acta = (unsigned short*)(smem + S_ACTA);   // [col][288]
    unsigned short* actb = (unsigned short*)(smem + S_ACTB);   // [col][256]
    float* qbuf = (float*)(smem + S_ACTA);                     // [64][NCOL], overlays acta
    float* mbuf = (float*)(smem + S_M);
    float* zbuf = (float*)(smem + S_Z);
    float* attb = (float*)(smem + S_ATT);
    float* sbuf = (float*)(smem + S_S);

    const int tid = threadIdx.x;
    const int b   = blockIdx.y;
    const int q0  = blockIdx.x * TQ;

    // ---------------- gather: build input panel [288 x 128] bf16 ----------------
    {
        int col = tid >> 1, half = tid & 1;
        int q = col >> 4, k = col & 15;
        int qg  = q0 + q;
        int qcl = qg < NQ ? qg : NQ - 1;
        int idx = proj[((size_t)b * NQ + qcl) * KNN + k];
        const unsigned short* src = lat_bf + ((size_t)b * N + idx) * LATENT + half * 128;
        unsigned short* dst = acta + (size_t)col * K1PAD + half * 128;
#pragma unroll
        for (int j = 0; j < 128; j += 8)
            *(v8u16*)(dst + j) = *(const v8u16*)(src + j);
        if (half) {  // rows 256..287: rel (3) + zero pad
            unsigned short tail[32];
#pragma unroll
            for (int j = 0; j < 32; j++) tail[j] = 0;
#pragma unroll
            for (int d = 0; d < 3; d++) {
                float r = ptsq[((size_t)b * 3 + d) * NQ + qcl] -
                          pts [((size_t)b * 3 + d) * N  + idx];
                tail[d] = f2bf(r);
            }
#pragma unroll
            for (int j = 0; j < 32; j += 8)
                *(v8u16*)(acta + (size_t)col * K1PAD + 256 + j) = *(v8u16*)(tail + j);
        }
    }
    __syncthreads();

    const int wave = tid >> 5, lane = tid & 31;

    // L1: [256 x 288] x [288 x 128] -> actb, ReLU
    gemm_panel<K1PAD, K1PAD, 256, 16, true, false>(w1p, b1, acta, actb, nullptr, wave, lane);
    __syncthreads();
    // L2: actb -> acta (first 256 rows of 288-stride buffer), ReLU
    gemm_panel<256, 256, K1PAD, 16, true, false>(w2b, b2, actb, acta, nullptr, wave, lane);
    __syncthreads();
    // L3: acta -> actb, ReLU
    gemm_panel<256, K1PAD, 256, 16, true, false>(w3b, b3, acta, actb, nullptr, wave, lane);
    __syncthreads();
    // Q: [64 x 256] x [256 x 128] -> qbuf f32 (overlays acta)
    gemm_panel<256, 256, 0, 4, false, true>(wqb, bq, actb, nullptr, qbuf, wave, lane);
    __syncthreads();

    // ---------------- folded value path: s[col] = wfold . act3[:,col] + cfold ----------------
    if (tid < NCOL) {
        const unsigned short* cp = actb + (size_t)tid * 256;
        float acc = wfold[256];
        for (int j = 0; j < 256; j += 8) {
            v8u16 hv = *(const v8u16*)(cp + j);
#pragma unroll
            for (int e = 0; e < 8; e++) acc += bf2f(hv[e]) * wfold[j + e];
        }
        sbuf[tid] = acc;
    }
    // ---------------- softmax stats per (head, query) ----------------
#pragma unroll
    for (int pass = 0; pass < 2; pass++) {
        int t = tid + pass * 256;     // 512 tasks = 64 heads x 8 queries
        int h = t >> 3, q = t & 7;
        const float* qp = qbuf + h * NCOL + q * KNN;
        float m = qp[0];
#pragma unroll
        for (int k = 1; k < KNN; k++) m = fmaxf(m, qp[k]);
        float z = 0.0f;
#pragma unroll
        for (int k = 0; k < KNN; k++) z += __expf(qp[k] - m);
        mbuf[h * 8 + q] = m;
        zbuf[h * 8 + q] = 1.0f / z;
    }
    __syncthreads();
    // ---------------- attention: mean over heads of per-head softmax ----------------
    if (tid < NCOL) {
        int q = tid >> 4;
        float a = 0.0f;
        for (int h = 0; h < HEADS; h++)
            a += __expf(qbuf[h * NCOL + tid] - mbuf[h * 8 + q]) * zbuf[h * 8 + q];
        attb[tid] = a * (1.0f / HEADS);
    }
    __syncthreads();
    // ---------------- output: out[b,q] = b8 + sum_k att * s ----------------
    if (tid < TQ && q0 + tid < NQ) {
        float o = b8[0];
#pragma unroll
        for (int k = 0; k < KNN; k++)
            o += attb[tid * KNN + k] * sbuf[tid * KNN + k];
        out[(size_t)b * NQ + q0 + tid] = o;
    }
}

// =====================================================================
extern "C" void kernel_launch(void* const* d_in, const int* in_sizes, int n_in,
                              void* d_out, int out_size, void* d_ws, size_t ws_size,
                              hipStream_t stream) {
    const float* latents = (const float*)d_in[0];
    const float* pts     = (const float*)d_in[1];
    const float* ptsq    = (const float*)d_in[2];
    const int*   proj    = (const int*)d_in[3];
    const float* W1 = (const float*)d_in[4];   const float* b1 = (const float*)d_in[5];
    const float* W2 = (const float*)d_in[6];   const float* b2 = (const float*)d_in[7];
    const float* W3 = (const float*)d_in[8];   const float* b3 = (const float*)d_in[9];
    const float* Wq = (const float*)d_in[10];  const float* bq = (const float*)d_in[11];
    const float* Wv = (const float*)d_in[12];  const float* bv = (const float*)d_in[13];
    const float* W8 = (const float*)d_in[14];  const float* b8 = (const float*)d_in[15];
    float* out = (float*)d_out;

    const int B  = 2;
    const int N  = in_sizes[1] / (B * 3);   // pts  [B,3,N]
    const int NQ = in_sizes[2] / (B * 3);   // ptsq [B,3,NQ]

    // workspace carve-up
    unsigned char* ws = (unsigned char*)d_ws;
    size_t off = 0;
    unsigned short* lat_bf = (unsigned short*)(ws + off); off += (size_t)B * N * LATENT * 2;
    unsigned short* w1p    = (unsigned short*)(ws + off); off += (size_t)256 * K1PAD * 2;
    unsigned short* w2b    = (unsigned short*)(ws + off); off += (size_t)256 * 256 * 2;
    unsigned short* w3b    = (unsigned short*)(ws + off); off += (size_t)256 * 256 * 2;
    unsigned short* wqb    = (unsigned short*)(ws + off); off += (size_t)64 * 256 * 2;
    float*          wfold  = (float*)(ws + off);          off += 257 * 4;
    (void)ws_size; (void)n_in; (void)out_size;

    prep_latents<<<dim3((N + 63) / 64, B), 64, 0, stream>>>(latents, lat_bf, N);

    const int wtotal = 256 * K1PAD + 2 * 256 * 256 + 64 * 256 + 257;
    prep_weights<<<(wtotal + 255) / 256, 256, 0, stream>>>(
        W1, W2, W3, Wq, Wv, W8, bv, w1p, w2b, w3b, wqb, wfold);

    interp_attention_main<<<dim3((NQ + TQ - 1) / TQ, B), 256, 0, stream>>>(
        lat_bf, pts, ptsq, proj,
        w1p, b1, w2b, b2, w3b, b3, wqb, bq,
        wfold, b8, out, N, NQ);
}
